// MultiAttn_25082609009273
// MI455X (gfx1250) — matrix operand order repaired and checked by reading, not compiled, not run on hardware
//
#include <hip/hip_runtime.h>

#define HEADS 4
#define BB 8
#define NN 1024
#define CC 256
#define DD 64

typedef __attribute__((ext_vector_type(16))) _Float16 v16h;
typedef __attribute__((ext_vector_type(8)))  float    v8f;

// ---- A-fragment (16x32 f16) from row-major f32 source (global or LDS) ----
// 16-bit A 16x32: lane l (0-15), hi=0: K = 0..15
//                 lane l+16 (hi=1): K = 16..31 (matches contiguous B fragments)
__device__ __forceinline__ v16h load_afrag_f32(const float* base, int stride, int l, int hi) {
    const float* r = base + (size_t)l * stride + hi * 16;
    float4 a = *(const float4*)(r);
    float4 b = *(const float4*)(r + 4);
    float4 c = *(const float4*)(r + 8);
    float4 d = *(const float4*)(r + 12);
    v16h o;
    o[0]=(_Float16)a.x; o[1]=(_Float16)a.y; o[2]=(_Float16)a.z; o[3]=(_Float16)a.w;
    o[4]=(_Float16)b.x; o[5]=(_Float16)b.y; o[6]=(_Float16)b.z; o[7]=(_Float16)b.w;
    o[8]=(_Float16)c.x; o[9]=(_Float16)c.y; o[10]=(_Float16)c.z; o[11]=(_Float16)c.w;
    o[12]=(_Float16)d.x; o[13]=(_Float16)d.y; o[14]=(_Float16)d.z; o[15]=(_Float16)d.w;
    return o;
}

// ---- B-fragment (32x16 f16) from f16 row-major source: lane l = column n,
// lanes 0-15 hold K=0..15, lanes 16-31 hold K=16..31 -> one contiguous 32B read.
__device__ __forceinline__ v16h load_bfrag_f16(const _Float16* base, int stride, int l, int hi) {
    return *(const v16h*)(base + (size_t)l * stride + hi * 16);
}

// ============ pre-pass: K -> f16 [b][h][n][d], V -> f16 transposed [b][h][d][n] ============
__global__ __launch_bounds__(256)
void prep_kernel(const float* __restrict__ k, const float* __restrict__ v,
                 _Float16* __restrict__ kf, _Float16* __restrict__ vt) {
    __shared__ _Float16 sh[64][72];  // padded transpose staging
    int t   = blockIdx.x;            // grid = BB*HEADS*(NN/64) = 512
    int nb  = t & 15;
    int bh  = t >> 4;
    int h   = bh & 3;
    int b   = bh >> 2;
    int n0  = nb * 64;
    int tid = threadIdx.x;
#pragma unroll
    for (int i = 0; i < 16; i++) {
        int lin = tid + 256 * i;
        int c = lin & 63;
        int n = lin >> 6;
        size_t src = ((size_t)(b * NN + n0 + n)) * CC + h * DD + c;
        // K: straight convert (row-major per head), coalesced in and out
        kf[((size_t)(b * HEADS + h) * NN + n0 + n) * DD + c] = (_Float16)k[src];
        // V: stage for transpose
        sh[c][n] = (_Float16)v[src];
    }
    __syncthreads();
#pragma unroll
    for (int i = 0; i < 16; i++) {
        int lin = tid + 256 * i;
        int n = lin & 63;
        int c = lin >> 6;
        vt[((size_t)(b * HEADS + h) * DD + c) * NN + n0 + n] = sh[c][n];
    }
}

// =====================  fused attention, one wave per 16-row tile  ===========
__global__ __launch_bounds__(32)
void attn_kernel(const float* __restrict__ q, const _Float16* __restrict__ kf,
                 const unsigned char* __restrict__ mask, const float* __restrict__ dis,
                 const _Float16* __restrict__ vt,
                 float* __restrict__ out, float* __restrict__ pattn) {
    __shared__ float sS[16 * NN];  // 64 KB score slab (one wave per workgroup)

    const int lane = threadIdx.x;
    const int l  = lane & 15;
    const int hi = lane >> 4;

    const int t    = blockIdx.x;       // grid = BB*HEADS*(NN/16) = 2048
    const int rt   = t & 63;
    const int bh   = t >> 6;           // bh = h*BB + b  (matches p_attn layout)
    const int b    = bh & 7;
    const int h    = bh >> 3;
    const int row0 = rt * 16;
    const int bh2  = b * HEADS + h;    // layout index for kf/vt

    // ---- resident Q A-fragments (16 rows x 64 ch, split K into 2x32) ----
    const float* qbase = q + ((size_t)(b * NN + row0)) * CC + h * DD;
    const v16h qa0 = load_afrag_f32(qbase,      CC, l, hi);
    const v16h qa1 = load_afrag_f32(qbase + 32, CC, l, hi);

    // ---- phase 1: scores = (Q K^T + dis)/8, masked, into LDS ----
    const _Float16* kslab = kf + (size_t)bh2 * NN * DD;
    for (int j = 0; j < 64; j++) {
        const _Float16* kb = kslab + (size_t)(j * 16) * DD + (size_t)l * DD + hi * 16;
        v16h kb0 = *(const v16h*)(kb);        // channels hi*16 .. +15
        v16h kb1 = *(const v16h*)(kb + 32);   // channels 32+hi*16 .. +15
        v8f c = {};
        c = __builtin_amdgcn_wmma_f32_16x16x32_f16(false, qa0, false, kb0, (short)0, c, false, false);
        c = __builtin_amdgcn_wmma_f32_16x16x32_f16(false, qa1, false, kb1, (short)0, c, false, false);
        // C layout: VGPR r, lanes 0-15 -> (M=r, N=l); lanes 16-31 -> (M=r+8, N=l)
        const float*         disb = dis  + ((size_t)(b * NN + row0 + hi * 8)) * NN + j * 16 + l;
        const unsigned char* mkb  = mask + ((size_t)(b * NN + row0 + hi * 8)) * NN + j * 16 + l;
#pragma unroll
        for (int r = 0; r < 8; r++) {
            float s = (c[r] + disb[(size_t)r * NN]) * 0.125f;   // 1/sqrt(64)
            if (mkb[(size_t)r * NN]) s = -1.0e30f;
            sS[(r + hi * 8) * NN + j * 16 + l] = s;
        }
    }

    // ---- phase 2: row softmax over LDS (unnormalized exp kept in LDS) ----
    float myRinv = 0.0f;  // lane m holds 1/rowsum of row m (m < 16)
    for (int m = 0; m < 16; m++) {
        float4* row4 = (float4*)(sS + (size_t)m * NN);
        float mx = -3.0e38f;
        for (int i = lane; i < NN / 4; i += 32) {
            float4 s4 = row4[i];
            mx = fmaxf(mx, fmaxf(fmaxf(s4.x, s4.y), fmaxf(s4.z, s4.w)));
        }
#pragma unroll
        for (int o = 16; o >= 1; o >>= 1) mx = fmaxf(mx, __shfl_xor(mx, o, 32));
        float sum = 0.0f;
        for (int i = lane; i < NN / 4; i += 32) {
            float4 s4 = row4[i];
            s4.x = __expf(s4.x - mx); s4.y = __expf(s4.y - mx);
            s4.z = __expf(s4.z - mx); s4.w = __expf(s4.w - mx);
            row4[i] = s4;
            sum += (s4.x + s4.y) + (s4.z + s4.w);
        }
#pragma unroll
        for (int o = 16; o >= 1; o >>= 1) sum += __shfl_xor(sum, o, 32);
        float rinv = 1.0f / sum;
        if (lane == m) myRinv = rinv;
    }

    // ---- phase 3: coalesced normalized p_attn store (float4) ----
    float* pbase = pattn + ((size_t)(bh * NN + row0)) * NN;
    for (int m = 0; m < 16; m++) {
        float rinv = __shfl(myRinv, m, 32);
        const float4* src = (const float4*)(sS + (size_t)m * NN);
        float4*       dst = (float4*)(pbase + (size_t)m * NN);
        for (int i = lane; i < NN / 4; i += 32) {
            float4 vv = src[i];
            vv.x *= rinv; vv.y *= rinv; vv.z *= rinv; vv.w *= rinv;
            dst[i] = vv;
        }
    }

    // ---- phase 4: O = P V  (P unnormalized in LDS; fold 1/sum at the end) ----
    v8f zero = {};
    v8f acc[4] = {zero, zero, zero, zero};
    const _Float16* vtb = vt + (size_t)bh2 * DD * NN;
    for (int kc = 0; kc < 32; kc++) {
        int n0 = kc * 32;
        v16h pa = load_afrag_f32(sS + n0, NN, l, hi);
#pragma unroll
        for (int c2 = 0; c2 < 4; c2++) {
            v16h vb = load_bfrag_f16(vtb + (size_t)(c2 * 16) * NN + n0, NN, l, hi);
            acc[c2] = __builtin_amdgcn_wmma_f32_16x16x32_f16(false, pa, false, vb, (short)0, acc[c2], false, false);
        }
    }

    // ---- scale rows by 1/sum and store out[b][row][h*64 + c] ----
    float fac[8];
#pragma unroll
    for (int r = 0; r < 8; r++) fac[r] = __shfl(myRinv, hi * 8 + r, 32);
    float* ob = out + ((size_t)(b * NN + row0 + hi * 8)) * CC + h * DD + l;
#pragma unroll
    for (int c2 = 0; c2 < 4; c2++) {
#pragma unroll
        for (int r = 0; r < 8; r++) {
            ob[(size_t)r * CC + c2 * 16] = acc[c2][r] * fac[r];
        }
    }
}

extern "C" void kernel_launch(void* const* d_in, const int* in_sizes, int n_in,
                              void* d_out, int out_size, void* d_ws, size_t ws_size,
                              hipStream_t stream) {
    const float*         q    = (const float*)d_in[0];
    const float*         k    = (const float*)d_in[1];
    const float*         v    = (const float*)d_in[2];
    const unsigned char* mask = (const unsigned char*)d_in[3];  // jax bool = 1 byte
    const float*         dis  = (const float*)d_in[4];

    float* out   = (float*)d_out;
    float* pattn = out + (size_t)BB * NN * CC;  // tuple: (out, p_attn) concatenated

    _Float16* kf = (_Float16*)d_ws;                                   // 4 MB
    _Float16* vt = kf + (size_t)BB * HEADS * NN * DD;                 // +4 MB

    prep_kernel<<<BB * HEADS * (NN / 64), 256, 0, stream>>>(k, v, kf, vt);
    attn_kernel<<<BB * HEADS * (NN / 16), 32, 0, stream>>>(q, kf, mask, dis, vt, out, pattn);
}
